// TopkSelfAttention_2963527434303
// MI455X (gfx1250) — matrix operand order repaired and checked by reading, not compiled, not run on hardware
//
#include <hip/hip_runtime.h>
#include <math.h>

// ---------------------------------------------------------------------------
// CDNA5 WMMA (wave32): v_wmma_f32_16x16x32_bf16 + Tensor Data Mover
// ---------------------------------------------------------------------------
typedef __bf16 bf16_t;
typedef __attribute__((ext_vector_type(16))) __bf16 v16bf;
typedef __attribute__((ext_vector_type(8)))  __bf16 v8bf;
typedef __attribute__((ext_vector_type(8)))  float  v8f;
typedef __attribute__((ext_vector_type(4)))  unsigned u32x4;
typedef __attribute__((ext_vector_type(8)))  int      i32x8;
typedef __attribute__((ext_vector_type(4)))  int      i32x4;

#define WMMA_BF16(a, b, c) \
  __builtin_amdgcn_wmma_f32_16x16x32_bf16(false, (a), false, (b), (short)0, (c), false, false)

#if defined(__has_builtin)
#  if __has_builtin(__builtin_amdgcn_tensor_load_to_lds) && \
      __has_builtin(__builtin_amdgcn_s_wait_tensorcnt)
#    define HAVE_TDM 1
#  endif
#endif
#ifndef HAVE_TDM
#  define HAVE_TDM 0
#endif

constexpr int Bb    = 8;
constexpr int Hh    = 8;
constexpr int Kk    = 256;
constexpr int Ntok  = 1024;
constexpr int PLANE = 4096;

__device__ __forceinline__ unsigned short f2bf_u(float f) {
  union { float f; unsigned u; } x; x.f = f;
  return (unsigned short)((x.u + 0x7FFFu + ((x.u >> 16) & 1u)) >> 16);  // RNE
}

// A fragment (16x32 MxK) from a ROW-MAJOR source: each lane's
// 16 elements are K {kb..kb+15} (two contiguous 16B runs),
// kb = 0 (lanes 0-15) / 16 (lanes 16-31). Two ds_load_b128 per fragment.
__device__ __forceinline__ v16bf loadA_rm(const bf16_t* src, int ld, int lane) {
  int m  = lane & 15;
  int kb = (lane >> 4) << 4;
  v8bf lo = *reinterpret_cast<const v8bf*>(src + m * ld + kb);
  v8bf hi = *reinterpret_cast<const v8bf*>(src + m * ld + kb + 8);
  return __builtin_shufflevector(lo, hi, 0, 1, 2, 3, 4, 5, 6, 7,
                                 8, 9, 10, 11, 12, 13, 14, 15);
}

// B fragment (32x16 KxN) from the fragment-swizzled image: 32B per lane.
__device__ __forceinline__ v16bf frag(const bf16_t* buf, int lane) {
  return *reinterpret_cast<const v16bf*>(buf + lane * 16);
}

__device__ __forceinline__ unsigned lds_off(const void* p) {
  // generic LDS address = {SHARED_BASE aperture, offset[31:0]} -> low 32 bits
  return (unsigned)(unsigned long long)(uintptr_t)p;
}

#if HAVE_TDM
// 2D (or 1D with tile_d1==0) bf16 tile load, global -> LDS, via TDM.
// Dims/strides in 2-byte elements. D# packing per CDNA5 ISA 8.3/8.4.
__device__ __forceinline__ void tdm_load_2d(unsigned lds_addr, const void* gptr,
                                            unsigned tile_d0, unsigned tile_d1,
                                            unsigned tensor_d0, unsigned tensor_d1,
                                            unsigned long long stride0) {
  unsigned long long ga = (unsigned long long)(uintptr_t)gptr;
  u32x4 g0;
  g0[0] = 1u;                                        // count=1, user descriptor
  g0[1] = lds_addr;                                  // LDS byte address
  g0[2] = (unsigned)ga;                              // global addr [31:0]
  g0[3] = (unsigned)(ga >> 32) | (2u << 30);         // addr [56:32] | type=2
  i32x8 g1;
  g1[0] = (int)(1u << 16);                           // data_size=1 (2 bytes)
  g1[1] = (int)((tensor_d0 & 0xFFFFu) << 16);        // tensor_dim0 lo16
  g1[2] = (int)((tensor_d0 >> 16) | ((tensor_d1 & 0xFFFFu) << 16));
  g1[3] = (int)((tensor_d1 >> 16) | (tile_d0 << 16));
  g1[4] = (int)(tile_d1 & 0xFFFFu);                  // tile_dim1 (0 => 1D)
  g1[5] = (int)(unsigned)(stride0 & 0xFFFFFFFFu);    // dim0 stride lo32
  g1[6] = (int)(unsigned)(stride0 >> 32);            // dim0 stride hi16
  g1[7] = 0;
  i32x4 z = {};
#if defined(__clang_major__) && (__clang_major__ >= 23)
  i32x8 z8 = {};
  __builtin_amdgcn_tensor_load_to_lds(g0, g1, z, z, z8, 0);
#else
  __builtin_amdgcn_tensor_load_to_lds(g0, g1, z, z, 0);
#endif
}
#endif

__device__ __forceinline__ float rowmax16(float v) {
  v = fmaxf(v, __shfl_xor(v, 1, 32));
  v = fmaxf(v, __shfl_xor(v, 2, 32));
  v = fmaxf(v, __shfl_xor(v, 4, 32));
  v = fmaxf(v, __shfl_xor(v, 8, 32));
  return v;
}
__device__ __forceinline__ float rowsum16(float v) {
  v += __shfl_xor(v, 1, 32);
  v += __shfl_xor(v, 2, 32);
  v += __shfl_xor(v, 4, 32);
  v += __shfl_xor(v, 8, 32);
  return v;
}

// ---------------------------------------------------------------------------
// Kernel 0: residual — permuted copy of x into d_out (one 2x2 patch / thread)
// ---------------------------------------------------------------------------
__global__ void residual_kernel(const float* __restrict__ x, float* __restrict__ out) {
  size_t i = (size_t)blockIdx.x * 256 + threadIdx.x;
  int p = (int)(i & 1023);
  size_t pl = i >> 10;
  int py = p >> 5, px = p & 31;
  const float* src = &x[pl * PLANE + (2 * py) * 64 + 2 * px];
  float2 r0 = *reinterpret_cast<const float2*>(src);
  float2 r1 = *reinterpret_cast<const float2*>(src + 64);
  *reinterpret_cast<float4*>(&out[pl * PLANE + p * 4]) = make_float4(r0.x, r0.y, r1.x, r1.y);
}

// ---------------------------------------------------------------------------
// Kernel 1: gather tokens + QKV GEMM.
//   qk[bh][tok][128] bf16 (cols 0-63 = Q, 64-127 = K)
//   vtws[bh][tb][8192] bf16 = V pre-swizzled into the B-fragment LDS image
// ---------------------------------------------------------------------------
__global__ void qkv_kernel(const float* __restrict__ x, const int* __restrict__ topk,
                           const float* __restrict__ Wq, const float* __restrict__ bq,
                           bf16_t* __restrict__ qk, bf16_t* __restrict__ vtws) {
  __shared__ __align__(32) bf16_t sT[128 * 64];       // token tile, row-major (16KB)
  __shared__ __align__(32) bf16_t sW[2 * 12 * 512];   // W_qkv, swizzled B     (24KB)

  int blk = blockIdx.x;
  int bh = blk >> 3, tb = blk & 7;
  int b = bh >> 3, hh = bh & 7;
  int tid = threadIdx.x;

  // W -> swizzled-B LDS (paired k rows, b32 stores)
  for (int pi = tid; pi < 32 * 192; pi += 256) {
    int nf = pi % 192;
    int k  = (pi / 192) * 2;
    float w0 = Wq[(size_t)k * 192 + nf];
    float w1 = Wq[(size_t)(k + 1) * 192 + nf];
    unsigned u = f2bf_u(w0) | ((unsigned)f2bf_u(w1) << 16);
    int ct = nf >> 4, nn = nf & 15;
    int kt = k >> 5, kk = k & 31;
    int L = nn | (((kk >> 4) & 1) << 4);
    *reinterpret_cast<unsigned*>(sW + (kt * 12 + ct) * 512 + L * 16 + (kk & 15)) = u;
  }

  // gather 128 tokens x 64 ch -> row-major sT (d-pairs, b32 stores)
  int d0 = (tid & 31) * 2;
  for (int it = 0; it < 16; ++it) {
    int nTok = (tid >> 5) + it * 8;
    int ng = tb * 128 + nTok;
    int kidx = ng >> 2, q2 = ng & 3;
    int p = topk[bh * Kk + kidx];
    int py = p >> 5, px = p & 31;
    int row = 2 * py + (q2 >> 1), col = 2 * px + (q2 & 1);
    const float* s0 = &x[((size_t)(b * 512 + hh * 64 + d0)) * PLANE + row * 64 + col];
    __builtin_prefetch(s0 + 2 * PLANE, 0, 1);
    unsigned u = f2bf_u(s0[0]) | ((unsigned)f2bf_u(s0[PLANE]) << 16);
    *reinterpret_cast<unsigned*>(sT + nTok * 64 + d0) = u;
  }
  __syncthreads();

  int wave = tid >> 5, lane = tid & 31;
  int n = lane & 15, hi8 = (lane >> 4) << 3;
  bool odd = lane & 1;
  int colBase = n & ~1;
  int tokBase = tb * 128 + wave * 16;

  v16bf a0 = loadA_rm(sT + wave * 16 * 64, 64, lane);
  v16bf a1 = loadA_rm(sT + wave * 16 * 64 + 32, 64, lane);

#pragma unroll
  for (int cp = 0; cp < 6; ++cp) {
    int c0 = cp * 2, c1 = c0 + 1;
    v16bf b00 = frag(sW + (0 * 12 + c0) * 512, lane);
    v16bf b01 = frag(sW + (1 * 12 + c0) * 512, lane);
    v16bf b10 = frag(sW + (0 * 12 + c1) * 512, lane);
    v16bf b11 = frag(sW + (1 * 12 + c1) * 512, lane);
    float bias0 = bq[c0 * 16 + n], bias1 = bq[c1 * 16 + n];
    v8f acc0, acc1;
#pragma unroll
    for (int v = 0; v < 8; ++v) { acc0[v] = bias0; acc1[v] = bias1; }
    acc0 = WMMA_BF16(a0, b00, acc0);
    acc0 = WMMA_BF16(a1, b01, acc0);
    acc1 = WMMA_BF16(a0, b10, acc1);
    acc1 = WMMA_BF16(a1, b11, acc1);

    if (c1 < 8) {
      int ctS = odd ? c1 : c0;
#pragma unroll
      for (int v = 0; v < 8; ++v) {
        float p0 = acc0[v], p1 = acc1[v];
        float q0 = __shfl_xor(p0, 1, 32);
        float q1 = __shfl_xor(p1, 1, 32);
        float lo = odd ? q1 : p0;
        float hi = odd ? p1 : q0;
        unsigned u = f2bf_u(lo) | ((unsigned)f2bf_u(hi) << 16);
        int tok = tokBase + v + hi8;
        *reinterpret_cast<unsigned*>(qk + ((size_t)bh * Ntok + tok) * 128 + ctS * 16 + colBase) = u;
      }
    } else {
      // V: write directly into the swizzled B-fragment image for this 128-token
      // block: chunk (kt = wave>>1, ct = c-8), lane L = n | ((wave&1)<<4),
      // elems (v+hi8) -> one b128 store per tile.
      int ktv = wave >> 1;
      int Lv = n | ((wave & 1) << 4);
      size_t vbase = ((size_t)(bh * 8 + tb)) * 8192;
#pragma unroll
      for (int cc = 0; cc < 2; ++cc) {
        v8f A = cc ? acc1 : acc0;
        int ct = (cc ? c1 : c0) - 8;
        uint4 pk;
        pk.x = f2bf_u(A[0]) | ((unsigned)f2bf_u(A[1]) << 16);
        pk.y = f2bf_u(A[2]) | ((unsigned)f2bf_u(A[3]) << 16);
        pk.z = f2bf_u(A[4]) | ((unsigned)f2bf_u(A[5]) << 16);
        pk.w = f2bf_u(A[6]) | ((unsigned)f2bf_u(A[7]) << 16);
        *reinterpret_cast<uint4*>(vtws + vbase + (ktv * 4 + ct) * 512 + Lv * 16 + hi8) = pk;
      }
    }
  }
}

// ---------------------------------------------------------------------------
// Kernel 2: flash attention (S = K·Qᵀ, softmax over m, O = P·V) + scatter-add.
// Tile fills via Tensor Data Mover (fallback: manual vector copies).
// ---------------------------------------------------------------------------
__global__ void attn_kernel(const bf16_t* __restrict__ qk, const bf16_t* __restrict__ vtws,
                            const int* __restrict__ topk, float* __restrict__ out) {
  __shared__ __align__(32) bf16_t sQ[128 * 64];     // row-major Q tile       (16KB)
  __shared__ __align__(32) bf16_t sV[16 * 512];     // swizzled-B V image     (16KB)
  __shared__ __align__(32) bf16_t sKP[8 * 2048];    // K stage, then per-wave P (32KB)

  int blk = blockIdx.x;
  int bh = blk >> 3, rb = blk & 7;
  int b = bh >> 3, hh = bh & 7;
  int tid = threadIdx.x, wave = tid >> 5, lane = tid & 31;
  int n = lane & 15, hi8 = (lane >> 4) << 3;
  int kb = (lane >> 4) << 4;
  bool odd = lane & 1;
  int colBase = n & ~1;

  // --- stage this block's 128 K-rows (row-major) ---
#if HAVE_TDM
  if (wave == 0) {
    tdm_load_2d(lds_off(sKP), qk + ((size_t)bh * Ntok + rb * 128) * 128 + 64,
                /*tile*/ 64, 128, /*tensor*/ 64, 128, /*stride*/ 128);
    __builtin_amdgcn_s_wait_tensorcnt(0);
  }
#else
  for (int c = tid; c < 2048; c += 256) {
    int tokL = c >> 4, cc = c & 15;
    *reinterpret_cast<unsigned long long*>(sKP + tokL * 64 + cc * 4) =
        *reinterpret_cast<const unsigned long long*>(
            qk + ((size_t)bh * Ntok + rb * 128 + tokL) * 128 + 64 + cc * 4);
  }
#endif
  __syncthreads();
  v16bf ka0 = loadA_rm(sKP + wave * 16 * 64, 64, lane);
  v16bf ka1 = loadA_rm(sKP + wave * 16 * 64 + 32, 64, lane);
  bf16_t* sPw = sKP + wave * 2048;   // row-major 16x128; barrier below separates uses

  float mrow[8], lrow[8];
  v8f o[4];
#pragma unroll
  for (int v = 0; v < 8; ++v) { mrow[v] = -INFINITY; lrow[v] = 0.f; }
#pragma unroll
  for (int ct = 0; ct < 4; ++ct) o[ct] = {};

  for (int mb = 0; mb < 8; ++mb) {
    __syncthreads();   // previous tiles consumed; ka frags already in registers
#if HAVE_TDM
    if (wave == 0) {
      tdm_load_2d(lds_off(sQ), qk + ((size_t)bh * Ntok + mb * 128) * 128,
                  64, 128, 64, 128, 128);
      tdm_load_2d(lds_off(sV), vtws + ((size_t)(bh * 8 + mb)) * 8192,
                  8192, 0, 8192, 1, 8192);   // 1D contiguous block
      __builtin_amdgcn_s_wait_tensorcnt(0);
    }
#else
    for (int c = tid; c < 2048; c += 256) {
      int tokL = c >> 4, cc = c & 15;
      *reinterpret_cast<unsigned long long*>(sQ + tokL * 64 + cc * 4) =
          *reinterpret_cast<const unsigned long long*>(
              qk + ((size_t)bh * Ntok + mb * 128 + tokL) * 128 + cc * 4);
    }
    {
      const uint4* src4 = reinterpret_cast<const uint4*>(vtws + ((size_t)(bh * 8 + mb)) * 8192);
      uint4* dst4 = reinterpret_cast<uint4*>(sV);
      for (int c = tid; c < 1024; c += 256) dst4[c] = src4[c];
    }
#endif
    __syncthreads();

    // S tile (16 x 128): Qᵀ B-fragments are contiguous 32B rows of sQ
    v8f s[8];
#pragma unroll
    for (int ct = 0; ct < 8; ++ct) {
      const bf16_t* qrow = sQ + (ct * 16 + n) * 64 + kb;
      v16bf b0 = *reinterpret_cast<const v16bf*>(qrow);
      v16bf b1 = *reinterpret_cast<const v16bf*>(qrow + 32);
      v8f acc = {};
      acc = WMMA_BF16(ka0, b0, acc);
      acc = WMMA_BF16(ka1, b1, acc);
      s[ct] = acc;
    }

    // Online softmax: s[ct] becomes P = exp(S - mnew)
#pragma unroll
    for (int v = 0; v < 8; ++v) {
      float m = s[0][v];
#pragma unroll
      for (int ct = 1; ct < 8; ++ct) m = fmaxf(m, s[ct][v]);
      m = rowmax16(m);
      float mnew = fmaxf(m, mrow[v]);
      float scale = __expf(mrow[v] - mnew);
      mrow[v] = mnew;
      lrow[v] *= scale;
#pragma unroll
      for (int ct = 0; ct < 4; ++ct) o[ct][v] *= scale;
#pragma unroll
      for (int ct = 0; ct < 8; ++ct) {
        float p = __expf(s[ct][v] - mnew);
        lrow[v] += p;
        s[ct][v] = p;
      }
    }

    // P -> per-wave row-major LDS: paired-column b32 stores
#pragma unroll
    for (int cp = 0; cp < 4; ++cp) {
      int ct0 = cp * 2, ct1 = ct0 + 1;
      int ctS = odd ? ct1 : ct0;
#pragma unroll
      for (int v = 0; v < 8; ++v) {
        float p0 = s[ct0][v], p1 = s[ct1][v];
        float q0 = __shfl_xor(p0, 1, 32);
        float q1 = __shfl_xor(p1, 1, 32);
        float lo = odd ? q1 : p0;
        float hi = odd ? p1 : q0;
        unsigned u = f2bf_u(lo) | ((unsigned)f2bf_u(hi) << 16);
        *reinterpret_cast<unsigned*>(sPw + (v + hi8) * 128 + ctS * 16 + colBase) = u;
      }
    }

    // O += P(16x128) @ V(128x64)
#pragma unroll
    for (int kt = 0; kt < 4; ++kt) {
      v16bf pa = loadA_rm(sPw + kt * 32, 128, lane);
#pragma unroll
      for (int ct = 0; ct < 4; ++ct) {
        v16bf vb = frag(sV + (kt * 4 + ct) * 512, lane);
        o[ct] = WMMA_BF16(pa, vb, o[ct]);
      }
    }
  }

  // Normalize, scatter-add into d_out (duplicate top-k patches accumulate)
#pragma unroll
  for (int v = 0; v < 8; ++v) {
    float inv = 1.f / rowsum16(lrow[v]);
    int tok = rb * 128 + wave * 16 + v + hi8;
    int kidx = tok >> 2, q2 = tok & 3;
    int p = topk[bh * Kk + kidx];
#pragma unroll
    for (int ct = 0; ct < 4; ++ct) {
      int d = ct * 16 + n;
      float* dst = &out[((size_t)(b * 512 + hh * 64 + d)) * PLANE + p * 4 + q2];
      unsafeAtomicAdd(dst, o[ct][v] * inv);
    }
  }
}

// ---------------------------------------------------------------------------
extern "C" void kernel_launch(void* const* d_in, const int* in_sizes, int n_in,
                              void* d_out, int out_size, void* d_ws, size_t ws_size,
                              hipStream_t stream) {
  const float* x    = (const float*)d_in[0];
  const int*   topk = (const int*)d_in[1];
  const float* Wq   = (const float*)d_in[2];
  const float* bq   = (const float*)d_in[3];
  float* out = (float*)d_out;
  // Workspace: qk = 64*1024*128*2 = 16 MB, vtws = 64*8*8192*2 = 8 MB
  bf16_t* qk   = (bf16_t*)d_ws;
  bf16_t* vtws = qk + (size_t)64 * Ntok * 128;

  residual_kernel<<<dim3(16384), dim3(256), 0, stream>>>(x, out);
  qkv_kernel<<<dim3(Bb * Hh * 8), dim3(256), 0, stream>>>(x, topk, Wq, bq, qk, vtws);
  attn_kernel<<<dim3(Bb * Hh * 8), dim3(256), 0, stream>>>(qk, vtws, topk, out);
}
